// MultiHeadedAttention_42803644072181
// MI455X (gfx1250) — matrix operand order repaired and checked by reading, compiled, not run on hardware
//
#include <hip/hip_runtime.h>
#include <hip/hip_bf16.h>
#include <math.h>

// Flip to 0 if the assembler rejects the async-load syntax.
#define ASYNC_COPY 1

// ---------------------------------------------------------------------------
// Problem constants
// ---------------------------------------------------------------------------
#define B_      8
#define S_      512
#define D_      1024
#define H_      8
#define DPH_    128
#define MAXREL  32
#define NREL    65
#define RELLD   72
#define BH_     (B_ * H_)
#define NTOK    (B_ * S_)
#define NRNN    (3 * B_ * 4 * S_)
#define SCALE_INV 0.08838834764831845f

typedef __attribute__((ext_vector_type(16))) _Float16 v16h;
typedef __attribute__((ext_vector_type(8)))  _Float16 h8;
typedef __attribute__((ext_vector_type(8)))  float    v8f;

union F16x16 { v16h v; h8 h2[2]; _Float16 e[16]; };
union F32x8  { v8f  v; float e[8]; };

#define WMMA_(a, b, c) __builtin_amdgcn_wmma_f32_16x16x32_f16(false, (a).v, false, (b).v, (short)0, (c).v, false, false)

// ---------------------------------------------------------------------------
// Shared GEMM parameter block
// ---------------------------------------------------------------------------
struct GP {
  const _Float16* A;     long long sAb;  int ldA;
  const _Float16* Aalt;  int ldAalt;
  const _Float16* Bt;    long long sBb;  int ldB;
  const _Float16* Btalt; int ldBalt;
  int M, N, K, K1;
  const float* bias; const float* bias2;
  const float* addend; long long sAddb; int ldAdd;   // added AFTER tanh
  float*    outF; long long sOutFb; int ldOut;
  _Float16* outH;
  int mode;      // 0 plain, 1 QKV scatter, 2 RNN scatter, 3 CTX scatter
  int do_tanh;
  const unsigned char* mask;
  _Float16* d0; _Float16* d1; _Float16* d2;
};

// Shared epilogue: one 16x16 accumulator, C layout: lane->col n=lane&15,
// VGPR r -> row r + 8*(lane>=16).
__device__ __forceinline__ void epilogue8(const GP& p, const F32x8& acc,
                                          int rb, int col, int z)
{
  if (col >= p.N) return;
  float bcol = 0.f;
  if (p.bias)  bcol += p.bias[col];
  if (p.bias2) bcol += p.bias2[col];
  #pragma unroll
  for (int r = 0; r < 8; r++) {
    int row = rb + r;
    if (row >= p.M) break;
    float c = acc.e[r] + bcol;
    if (p.do_tanh) c = tanhf(c);
    if (p.addend)
      c += p.addend[(long long)z * p.sAddb + (long long)row * p.ldAdd + col];
    switch (p.mode) {
      case 0: {
        long long o = (long long)z * p.sOutFb + (long long)row * p.ldOut + col;
        if (p.outF) p.outF[o] = c;
        if (p.outH) p.outH[o] = (_Float16)c;
      } break;
      case 1: { // QKV: row=(b,s), col=(h,d) -> masked f16 (B,H,S,DPH)
        int b = row >> 9, s = row & 511;
        int h = col >> 7, d = col & 127;
        if (p.mask && p.mask[b * S_ + s]) c = 0.f;
        p.d0[(((long long)(b * H_ + h) * S_ + s) << 7) + d] = (_Float16)c;
      } break;
      case 2: { // RNN: row=(st,b,hh,s), col=d -> heads 4..7
        int st = row >> 14;
        int rem = row & 16383;
        int b = rem >> 11, hh = (rem >> 9) & 3, s = rem & 511;
        long long base = (long long)(b * H_ + 4 + hh);
        if (st == 0)      p.d0[((base * S_ + s) << 7) + col] = (_Float16)c;
        else if (st == 1) p.d1[((base * S_ + s) << 7) + col] = (_Float16)c;
        else              p.d2[((base << 7) + col) * S_ + s] = (_Float16)c;
      } break;
      case 3: { // CTX: z=bh, row=q, col=d -> ctx f16 (B,S,H,DPH)
        int b = z >> 3, h = z & 7;
        p.d0[(((long long)(b * S_ + row) * H_ + h) << 7) + col] = (_Float16)c;
      } break;
    }
  }
}

// ---------------------------------------------------------------------------
// FAST GEMM: exact tiles (M%128==0, N%64==0, K%64==0, K1%64==0).
// 128x64 block tile, 8 waves in 4x2 grid, each wave owns 2x2 16x16 subtiles.
// K staged 64 at a time in LDS -> 8 v_wmma per barrier pair.
// A/B staging uses global_load_async_to_lds_b128 (ASYNCcnt) when enabled.
// ---------------------------------------------------------------------------
#define FTM 128
#define FTN 64
#define FLD 72    // 64 + 8 pad halves -> 144B rows (16B aligned)

__device__ __forceinline__ void async_cp16(_Float16* lds_dst, const _Float16* gsrc)
{
#if ASYNC_COPY
  unsigned lo = (unsigned)(unsigned long long)(void*)lds_dst;
  asm volatile("global_load_async_to_lds_b128 %0, %1, off"
               :: "v"(lo), "v"(gsrc) : "memory");
#else
  *(h8*)lds_dst = *(const h8*)gsrc;
#endif
}

__global__ __launch_bounds__(256) void k_gemm_f(GP p)
{
  __shared__ _Float16 sA[FTM * FLD];
  __shared__ _Float16 sB[FTN * FLD];
  const int tid  = threadIdx.x;
  const int lane = tid & 31;
  const int wave = tid >> 5;
  const int mg = (wave >> 1) << 5;   // 0,32,64,96
  const int ng = (wave & 1) << 5;    // 0,32
  const int mblk = blockIdx.y * FTM;
  const int nblk = blockIdx.x * FTN;
  const int z    = blockIdx.z;

  const int ar = tid >> 1;           // 0..127 (A row)
  const int ac = (tid & 1) << 5;     // 0,32   (A col halves, 64B per thread)
  const int br = tid >> 2;           // 0..63  (B row)
  const int bc = (tid & 3) << 4;     // 0..48  (B col halves, 32B per thread)

  F32x8 acc[2][2];
  #pragma unroll
  for (int i = 0; i < 2; i++)
    #pragma unroll
    for (int j = 0; j < 2; j++)
      #pragma unroll
      for (int e = 0; e < 8; e++) acc[i][j].e[e] = 0.f;

  for (int k0 = 0; k0 < p.K; k0 += 64) {
    // uniform (SALU) segment select; K1 multiple of 64 so chunks never straddle
    const _Float16 *As, *Bs; int lA, lB, kk;
    if (k0 >= p.K1) {
      As = p.Aalt; lA = p.ldAalt; Bs = p.Btalt; lB = p.ldBalt; kk = k0 - p.K1;
    } else {
      As = p.A  + (long long)z * p.sAb; lA = p.ldA;
      Bs = p.Bt + (long long)z * p.sBb; lB = p.ldB;
      kk = k0;
    }
    __syncthreads();   // previous fragments consumed
    {
      const _Float16* ga = As + (long long)(mblk + ar) * lA + kk + ac;
      _Float16* da = sA + ar * FLD + ac;
#if ASYNC_COPY
      unsigned lo = (unsigned)(unsigned long long)(void*)da;
      asm volatile("global_load_async_to_lds_b128 %0, %1, off"           :: "v"(lo), "v"(ga) : "memory");
      asm volatile("global_load_async_to_lds_b128 %0, %1, off offset:16" :: "v"(lo), "v"(ga) : "memory");
      asm volatile("global_load_async_to_lds_b128 %0, %1, off offset:32" :: "v"(lo), "v"(ga) : "memory");
      asm volatile("global_load_async_to_lds_b128 %0, %1, off offset:48" :: "v"(lo), "v"(ga) : "memory");
#else
      #pragma unroll
      for (int c = 0; c < 4; c++) *(h8*)(da + 8 * c) = *(const h8*)(ga + 8 * c);
#endif
      if (k0 + 64 < p.K) __builtin_prefetch(ga + 64, 0, 1);
    }
    {
      const _Float16* gb = Bs + (long long)(nblk + br) * lB + kk + bc;
      _Float16* db = sB + br * FLD + bc;
#if ASYNC_COPY
      unsigned lo = (unsigned)(unsigned long long)(void*)db;
      asm volatile("global_load_async_to_lds_b128 %0, %1, off"           :: "v"(lo), "v"(gb) : "memory");
      asm volatile("global_load_async_to_lds_b128 %0, %1, off offset:16" :: "v"(lo), "v"(gb) : "memory");
#else
      #pragma unroll
      for (int c = 0; c < 2; c++) *(h8*)(db + 8 * c) = *(const h8*)(gb + 8 * c);
#endif
      if (k0 + 64 < p.K) __builtin_prefetch(gb + 64, 0, 1);
    }
#if ASYNC_COPY
    asm volatile("s_wait_asynccnt 0" ::: "memory");
#endif
    __syncthreads();

    const int al = lane & 15;
    const int ah = lane >> 4;
    #pragma unroll
    for (int kc = 0; kc < 64; kc += 32) {
      F16x16 a0, a1, b0, b1;
      // A 16x32 frag: lane row m=al, halves {ka..ka+7, ka+16..ka+23}, ka=8*ah
      const _Float16* ap0 = sA + (mg + al) * FLD + kc + (ah << 3);
      const _Float16* ap1 = ap0 + 16 * FLD;
      a0.h2[0] = *(const h8*)ap0;  a0.h2[1] = *(const h8*)(ap0 + 16);
      a1.h2[0] = *(const h8*)ap1;  a1.h2[1] = *(const h8*)(ap1 + 16);
      // B 32x16 frag: lane col n=al, halves {kb..kb+7, kb+16..kb+23}, kb=8*ah
      const _Float16* bp0 = sB + (ng + al) * FLD + kc + (ah << 3);
      const _Float16* bp1 = bp0 + 16 * FLD;
      b0.h2[0] = *(const h8*)bp0;  b0.h2[1] = *(const h8*)(bp0 + 16);
      b1.h2[0] = *(const h8*)bp1;  b1.h2[1] = *(const h8*)(bp1 + 16);

      acc[0][0].v = WMMA_(a0, b0, acc[0][0]);
      acc[0][1].v = WMMA_(a0, b1, acc[0][1]);
      acc[1][0].v = WMMA_(a1, b0, acc[1][0]);
      acc[1][1].v = WMMA_(a1, b1, acc[1][1]);
    }
  }
  __syncthreads();

  const int al = lane & 15;
  const int rhalf = (lane >> 4) << 3;
  #pragma unroll
  for (int ms = 0; ms < 2; ms++)
    #pragma unroll
    for (int ns = 0; ns < 2; ns++)
      epilogue8(p, acc[ms][ns],
                mblk + mg + ms * 16 + rhalf,
                nblk + ng + ns * 16 + al, z);
}

// ---------------------------------------------------------------------------
// GUARDED GEMM (ragged shapes: qrel N=65, ctx2 K=72). 64x32 tile, 8 waves.
// ---------------------------------------------------------------------------
#define GTM 64
#define GTN 32
#define GLDA 40
#define GLDB 40

__device__ __forceinline__ v8f gemm_core_g(const GP& p, int mblk, int nblk, int z,
                                           _Float16* sA, _Float16* sB)
{
  const int tid  = threadIdx.x;
  const int lane = tid & 31;
  const int wave = tid >> 5;
  const int mi = (wave >> 1) << 4;
  const int nj = (wave & 1) << 4;
  const _Float16* Abase = p.A  + (long long)z * p.sAb;
  const _Float16* Bbase = p.Bt + (long long)z * p.sBb;

  F32x8 acc;
  #pragma unroll
  for (int i = 0; i < 8; i++) acc.e[i] = 0.f;

  for (int k0 = 0; k0 < p.K; k0 += 32) {
    __syncthreads();
    {
      int r = tid >> 2, c = (tid & 3) << 3;
      int gm = mblk + r, gk = k0 + c;
      const _Float16* src; int ld, kk, kend;
      if (gk >= p.K1) { src = p.Aalt; ld = p.ldAalt; kk = gk - p.K1; kend = p.K - p.K1; }
      else            { src = Abase;  ld = p.ldA;    kk = gk;        kend = p.K1; }
      _Float16* dst = sA + r * GLDA + c;
      if (gm < p.M && (kk + 8) <= kend) {
        *(h8*)dst = *(const h8*)(src + (long long)gm * ld + kk);
      } else {
        #pragma unroll
        for (int i = 0; i < 8; i++) {
          _Float16 val = (_Float16)0.f;
          if (gm < p.M && (kk + i) < kend) val = src[(long long)gm * ld + kk + i];
          dst[i] = val;
        }
      }
    }
    if (tid < 128) {
      int r = tid >> 2, c = (tid & 3) << 3;
      int gn = nblk + r, gk = k0 + c;
      const _Float16* src; int ld, kk, kend;
      if (gk >= p.K1) { src = p.Btalt; ld = p.ldBalt; kk = gk - p.K1; kend = p.K - p.K1; }
      else            { src = Bbase;   ld = p.ldB;    kk = gk;        kend = p.K1; }
      _Float16* dst = sB + r * GLDB + c;
      if (gn < p.N && (kk + 8) <= kend) {
        *(h8*)dst = *(const h8*)(src + (long long)gn * ld + kk);
      } else {
        #pragma unroll
        for (int i = 0; i < 8; i++) {
          _Float16 val = (_Float16)0.f;
          if (gn < p.N && (kk + i) < kend) val = src[(long long)gn * ld + kk + i];
          dst[i] = val;
        }
      }
    }
    __syncthreads();
    F16x16 a, b;
    {
      int arow = mi + (lane & 15);
      int ka   = (lane >> 4) << 3;
      const _Float16* ap = sA + arow * GLDA + ka;
      a.h2[0] = *(const h8*)ap;
      a.h2[1] = *(const h8*)(ap + 16);
    }
    {
      int brow = nj + (lane & 15);
      int kb   = (lane >> 4) << 3;
      const _Float16* bp = sB + brow * GLDB + kb;
      b.h2[0] = *(const h8*)bp;
      b.h2[1] = *(const h8*)(bp + 16);
    }
    acc.v = WMMA_(a, b, acc);
  }
  __syncthreads();
  return acc.v;
}

__global__ __launch_bounds__(256) void k_gemm_g(GP p)
{
  __shared__ _Float16 sA[GTM * GLDA];
  __shared__ _Float16 sB[GTN * GLDB];
  const int mblk = blockIdx.y * GTM;
  const int nblk = blockIdx.x * GTN;
  const int z    = blockIdx.z;
  F32x8 acc; acc.v = gemm_core_g(p, mblk, nblk, z, sA, sB);
  const int lane = threadIdx.x & 31;
  const int wave = threadIdx.x >> 5;
  const int mi = (wave >> 1) << 4;
  const int nj = (wave & 1) << 4;
  epilogue8(p, acc, mblk + mi + ((lane >> 4) << 3), nblk + nj + (lane & 15), z);
}

// ---------------------------------------------------------------------------
// Small helper kernels
// ---------------------------------------------------------------------------
__global__ void k_cvt(const float* __restrict__ src, _Float16* __restrict__ dst, int n)
{
  int i = blockIdx.x * blockDim.x + threadIdx.x;
  if (i < n) dst[i] = (_Float16)src[i];
}

__global__ void k_relT(const float* __restrict__ rel, _Float16* __restrict__ relT)
{
  int i = blockIdx.x * blockDim.x + threadIdx.x;
  if (i >= DPH_ * RELLD) return;
  int d = i / RELLD, j = i % RELLD;
  relT[i] = (j < NREL) ? (_Float16)rel[j * DPH_ + d] : (_Float16)0.f;
}

__global__ void k_vT(const _Float16* __restrict__ v, _Float16* __restrict__ vT)
{
  int bh = blockIdx.y;
  int i  = blockIdx.x * blockDim.x + threadIdx.x;
  int s = i >> 7, d = i & 127;
  vT[((long long)bh * DPH_ + d) * S_ + s] = v[((long long)bh * S_ + s) * DPH_ + d];
}

__global__ void k_feats(const _Float16* __restrict__ q, const _Float16* __restrict__ k,
                        const _Float16* __restrict__ v,
                        _Float16* __restrict__ feats, _Float16* __restrict__ fshift)
{
  int i = blockIdx.x * blockDim.x + threadIdx.x;
  int d = i & 127;
  int n = i >> 7;
  int st = n >> 14, rem = n & 16383;
  int b = rem >> 11, hh = (rem >> 9) & 3, s = rem & 511;
  const _Float16* src = (st == 0) ? q : (st == 1) ? k : v;
  long long base = (long long)(b * H_ + 4 + hh) * S_;
  feats[i]  = src[((base + s) << 7) + d];
  fshift[i] = (s == 0) ? (_Float16)0.f : src[((base + s - 1) << 7) + d];
}

// ---------------------------------------------------------------------------
// rel + scale + mask + softmax + bucket sums
// ---------------------------------------------------------------------------
__device__ __forceinline__ float block_allreduce(float v, bool is_max,
                                                 float* sred, float* sbc)
{
  int lane = threadIdx.x & 31, wave = threadIdx.x >> 5;
  #pragma unroll
  for (int off = 16; off > 0; off >>= 1) {
    float o = __shfl_xor(v, off, 32);
    v = is_max ? fmaxf(v, o) : (v + o);
  }
  if (lane == 0) sred[wave] = v;
  __syncthreads();
  if (threadIdx.x == 0) {
    float r = sred[0];
    for (int w = 1; w < 8; w++) r = is_max ? fmaxf(r, sred[w]) : (r + sred[w]);
    *sbc = r;
  }
  __syncthreads();
  float r = *sbc;
  __syncthreads();
  return r;
}

__global__ __launch_bounds__(256) void k_softmax(const float* __restrict__ scores,
                                                 const float* __restrict__ qrel,
                                                 const unsigned char* __restrict__ mask,
                                                 float* __restrict__ attnF,
                                                 _Float16* __restrict__ attnH,
                                                 _Float16* __restrict__ asumH)
{
  const int q = blockIdx.x, bh = blockIdx.y, b = bh >> 3;
  const long long row = (long long)bh * S_ + q;
  const float* srow = scores + row * S_;
  __shared__ float sQr[NREL];
  __shared__ float sAs[RELLD];
  __shared__ float sred[8];
  __shared__ float sbc;
  const int tid = threadIdx.x;
  if (tid < NREL)  sQr[tid] = qrel[row * NREL + tid];
  if (tid < RELLD) sAs[tid] = 0.f;
  __syncthreads();

  float vals[2]; int bidx[2];
  float mx = -3.0e38f;
  #pragma unroll
  for (int i = 0; i < 2; i++) {
    int kk = tid + (i << 8);
    int dlt = kk - q;
    dlt = dlt < -MAXREL ? -MAXREL : (dlt > MAXREL ? MAXREL : dlt);
    bidx[i] = dlt + MAXREL;
    float s = (srow[kk] + sQr[bidx[i]]) * SCALE_INV;
    if (mask[b * S_ + kk]) s = -1.0e18f;
    vals[i] = s;
    mx = fmaxf(mx, s);
  }
  float m = block_allreduce(mx, true, sred, &sbc);
  float lsum = 0.f;
  #pragma unroll
  for (int i = 0; i < 2; i++) { vals[i] = __expf(vals[i] - m); lsum += vals[i]; }
  float tot = block_allreduce(lsum, false, sred, &sbc);
  float rinv = 1.f / tot;
  #pragma unroll
  for (int i = 0; i < 2; i++) {
    int kk = tid + (i << 8);
    float a = vals[i] * rinv;
    attnF[row * S_ + kk] = a;
    attnH[row * S_ + kk] = (_Float16)a;
    atomicAdd(&sAs[bidx[i]], a);
  }
  __syncthreads();
  if (tid < RELLD)
    asumH[row * RELLD + tid] = (tid < NREL) ? (_Float16)sAs[tid] : (_Float16)0.f;
}

// ---------------------------------------------------------------------------
// Host orchestration
// ---------------------------------------------------------------------------
extern "C" void kernel_launch(void* const* d_in, const int* in_sizes, int n_in,
                              void* d_out, int out_size, void* d_ws, size_t ws_size,
                              hipStream_t stream)
{
  (void)in_sizes; (void)n_in; (void)out_size; (void)ws_size;
  const float* x            = (const float*)d_in[0];
  const unsigned char* mask = (const unsigned char*)d_in[1];
  const float* Wq  = (const float*)d_in[2];
  const float* bq  = (const float*)d_in[3];
  const float* Wk  = (const float*)d_in[4];
  const float* bk  = (const float*)d_in[5];
  const float* Wv  = (const float*)d_in[6];
  const float* bv  = (const float*)d_in[7];
  const float* Wo  = (const float*)d_in[8];
  const float* bo  = (const float*)d_in[9];
  const float* rel = (const float*)d_in[10];
  const float* Wih_f = (const float*)d_in[11];
  const float* Whh_f = (const float*)d_in[12];
  const float* bih_f = (const float*)d_in[13];
  const float* bhh_f = (const float*)d_in[14];
  const float* Wih_b = (const float*)d_in[15];
  const float* Whh_b = (const float*)d_in[16];
  const float* bih_b = (const float*)d_in[17];
  const float* bhh_b = (const float*)d_in[18];

  float* outp  = (float*)d_out;
  float* attnF = (float*)d_out + (long long)B_ * S_ * D_;

  char* w = (char*)d_ws;
  auto alloc = [&](size_t bytes) -> char* {
    char* p = w; w += (bytes + 255) & ~(size_t)255; return p;
  };
  _Float16* xh     = (_Float16*)alloc((size_t)NTOK * D_ * 2);
  _Float16* Wqh    = (_Float16*)alloc((size_t)D_ * D_ * 2);
  _Float16* Wkh    = (_Float16*)alloc((size_t)D_ * D_ * 2);
  _Float16* Wvh    = (_Float16*)alloc((size_t)D_ * D_ * 2);
  _Float16* Woh    = (_Float16*)alloc((size_t)D_ * D_ * 2);
  _Float16* Wihf_h = (_Float16*)alloc((size_t)DPH_ * DPH_ * 2);
  _Float16* Whhf_h = (_Float16*)alloc((size_t)DPH_ * DPH_ * 2);
  _Float16* Wihb_h = (_Float16*)alloc((size_t)DPH_ * DPH_ * 2);
  _Float16* Whhb_h = (_Float16*)alloc((size_t)DPH_ * DPH_ * 2);
  _Float16* relh   = (_Float16*)alloc((size_t)NREL * DPH_ * 2);
  _Float16* relT16 = (_Float16*)alloc((size_t)DPH_ * RELLD * 2);
  _Float16* qh16   = (_Float16*)alloc((size_t)BH_ * S_ * DPH_ * 2);
  _Float16* kh16   = (_Float16*)alloc((size_t)BH_ * S_ * DPH_ * 2);
  _Float16* vh16   = (_Float16*)alloc((size_t)BH_ * S_ * DPH_ * 2);
  _Float16* vT16   = (_Float16*)alloc((size_t)BH_ * DPH_ * S_ * 2);
  _Float16* feats  = (_Float16*)alloc((size_t)NRNN * DPH_ * 2);
  _Float16* fshift = (_Float16*)alloc((size_t)NRNN * DPH_ * 2);
  _Float16* Pf16   = (_Float16*)alloc((size_t)NRNN * DPH_ * 2);
  _Float16* Pb16   = (_Float16*)alloc((size_t)NRNN * DPH_ * 2);
  float*    Hf32   = (float*)   alloc((size_t)NRNN * DPH_ * 4);
  float*    qrelb  = (float*)   alloc((size_t)BH_ * S_ * NREL * 4);
  float*    scores = (float*)   alloc((size_t)BH_ * S_ * S_ * 4);
  _Float16* attnH  = (_Float16*)alloc((size_t)BH_ * S_ * S_ * 2);
  _Float16* asumH  = (_Float16*)alloc((size_t)BH_ * S_ * RELLD * 2);
  float*    ctx1   = (float*)   alloc((size_t)BH_ * S_ * DPH_ * 4);
  _Float16* ctxh   = (_Float16*)alloc((size_t)NTOK * D_ * 2);

  auto cvt = [&](const float* s, _Float16* d, int n) {
    k_cvt<<<(n + 255) / 256, 256, 0, stream>>>(s, d, n);
  };
  cvt(x, xh, NTOK * D_);
  cvt(Wq, Wqh, D_ * D_);  cvt(Wk, Wkh, D_ * D_);
  cvt(Wv, Wvh, D_ * D_);  cvt(Wo, Woh, D_ * D_);
  cvt(Wih_f, Wihf_h, DPH_ * DPH_); cvt(Whh_f, Whhf_h, DPH_ * DPH_);
  cvt(Wih_b, Wihb_h, DPH_ * DPH_); cvt(Whh_b, Whhb_h, DPH_ * DPH_);
  cvt(rel, relh, NREL * DPH_);
  k_relT<<<(DPH_ * RELLD + 255) / 256, 256, 0, stream>>>(rel, relT16);

  GP g;
  auto def = [&](GP& p) { p = GP{}; };

  // ---- 1: QKV projections ----
  const float* biases[3] = { bq, bk, bv };
  const _Float16* wqkv[3] = { Wqh, Wkh, Wvh };
  _Float16* dsts[3] = { qh16, kh16, vh16 };
  for (int t = 0; t < 3; t++) {
    def(g);
    g.A = xh; g.ldA = D_; g.Aalt = xh; g.ldAalt = D_;
    g.Bt = wqkv[t]; g.ldB = D_; g.Btalt = wqkv[t]; g.ldBalt = D_;
    g.M = NTOK; g.N = D_; g.K = D_; g.K1 = D_;
    g.bias = biases[t]; g.mask = mask; g.mode = 1; g.d0 = dsts[t];
    k_gemm_f<<<dim3(D_ / FTN, NTOK / FTM, 1), 256, 0, stream>>>(g);
  }

  // ---- 2: v transpose for ctx GEMM B operand ----
  k_vT<<<dim3(256, BH_, 1), 256, 0, stream>>>(vh16, vT16);

  // ---- 3: RNN inputs ----
  k_feats<<<(NRNN * DPH_) / 256, 256, 0, stream>>>(qh16, kh16, vh16, feats, fshift);

  // ---- 4: RNN as 4 fused GEMMs ----
  def(g);
  g.A = fshift; g.ldA = DPH_; g.Aalt = fshift; g.ldAalt = DPH_;
  g.Bt = Wihf_h; g.ldB = DPH_; g.Btalt = Wihf_h; g.ldBalt = DPH_;
  g.M = NRNN; g.N = DPH_; g.K = DPH_; g.K1 = DPH_;
  g.bias = bih_f; g.bias2 = bhh_f; g.do_tanh = 1; g.outH = Pf16; g.ldOut = DPH_;
  k_gemm_f<<<dim3(DPH_ / FTN, NRNN / FTM, 1), 256, 0, stream>>>(g);

  g.A = feats; g.Aalt = feats; g.Bt = Wihb_h; g.Btalt = Wihb_h;
  g.bias = bih_b; g.bias2 = bhh_b; g.outH = Pb16;
  k_gemm_f<<<dim3(DPH_ / FTN, NRNN / FTM, 1), 256, 0, stream>>>(g);

  def(g);
  g.A = feats; g.ldA = DPH_; g.Aalt = Pf16; g.ldAalt = DPH_;
  g.Bt = Wihf_h; g.ldB = DPH_; g.Btalt = Whhf_h; g.ldBalt = DPH_;
  g.M = NRNN; g.N = DPH_; g.K = 2 * DPH_; g.K1 = DPH_;
  g.bias = bih_f; g.bias2 = bhh_f; g.do_tanh = 1; g.outF = Hf32; g.ldOut = DPH_;
  k_gemm_f<<<dim3(DPH_ / FTN, NRNN / FTM, 1), 256, 0, stream>>>(g);

  def(g);
  g.A = fshift; g.ldA = DPH_; g.Aalt = Pb16; g.ldAalt = DPH_;
  g.Bt = Wihb_h; g.ldB = DPH_; g.Btalt = Whhb_h; g.ldBalt = DPH_;
  g.M = NRNN; g.N = DPH_; g.K = 2 * DPH_; g.K1 = DPH_;
  g.bias = bih_b; g.bias2 = bhh_b; g.do_tanh = 1;
  g.addend = Hf32; g.ldAdd = DPH_;
  g.mode = 2; g.d0 = qh16; g.d1 = kh16; g.d2 = vT16;
  k_gemm_f<<<dim3(DPH_ / FTN, NRNN / FTM, 1), 256, 0, stream>>>(g);

  // ---- 5: qrel = q @ rel^T (ragged N=65 -> guarded kernel) ----
  def(g);
  g.A = qh16; g.ldA = DPH_; g.Aalt = qh16; g.ldAalt = DPH_;
  g.Bt = relh; g.ldB = DPH_; g.Btalt = relh; g.ldBalt = DPH_;
  g.M = BH_ * S_; g.N = NREL; g.K = DPH_; g.K1 = DPH_;
  g.outF = qrelb; g.ldOut = NREL;
  k_gemm_g<<<dim3((NREL + GTN - 1) / GTN, (BH_ * S_) / GTM, 1), 256, 0, stream>>>(g);

  // ---- 6: scores = q @ k^T (batched) ----
  def(g);
  g.A = qh16; g.sAb = (long long)S_ * DPH_; g.ldA = DPH_;
  g.Aalt = qh16; g.ldAalt = DPH_;
  g.Bt = kh16; g.sBb = (long long)S_ * DPH_; g.ldB = DPH_;
  g.Btalt = kh16; g.ldBalt = DPH_;
  g.M = S_; g.N = S_; g.K = DPH_; g.K1 = DPH_;
  g.outF = scores; g.sOutFb = (long long)S_ * S_; g.ldOut = S_;
  k_gemm_f<<<dim3(S_ / FTN, S_ / FTM, BH_), 256, 0, stream>>>(g);

  // ---- 7: softmax ----
  k_softmax<<<dim3(S_, BH_, 1), 256, 0, stream>>>(scores, qrelb, mask,
                                                  attnF, attnH, asumH);

  // ---- 8: ctx1 = attn @ v (batched) ----
  def(g);
  g.A = attnH; g.sAb = (long long)S_ * S_; g.ldA = S_;
  g.Aalt = attnH; g.ldAalt = S_;
  g.Bt = vT16; g.sBb = (long long)DPH_ * S_; g.ldB = S_;
  g.Btalt = vT16; g.ldBalt = S_;
  g.M = S_; g.N = DPH_; g.K = S_; g.K1 = S_;
  g.outF = ctx1; g.sOutFb = (long long)S_ * DPH_; g.ldOut = DPH_;
  k_gemm_f<<<dim3(DPH_ / FTN, S_ / FTM, BH_), 256, 0, stream>>>(g);

  // ---- 9: ctx = ctx1 + asum @ rel (ragged K -> guarded kernel) ----
  def(g);
  g.A = asumH; g.sAb = (long long)S_ * RELLD; g.ldA = RELLD;
  g.Aalt = asumH; g.ldAalt = RELLD;
  g.Bt = relT16; g.sBb = 0; g.ldB = RELLD;
  g.Btalt = relT16; g.ldBalt = RELLD;
  g.M = S_; g.N = DPH_; g.K = RELLD; g.K1 = RELLD;
  g.addend = ctx1; g.sAddb = (long long)S_ * DPH_; g.ldAdd = DPH_;
  g.mode = 3; g.d0 = ctxh;
  k_gemm_g<<<dim3(DPH_ / GTN, S_ / GTM, BH_), 256, 0, stream>>>(g);

  // ---- 10: out = ctx2d @ Wo^T + bo ----
  def(g);
  g.A = ctxh; g.ldA = D_; g.Aalt = ctxh; g.ldAalt = D_;
  g.Bt = Woh; g.ldB = D_; g.Btalt = Woh; g.ldBalt = D_;
  g.M = NTOK; g.N = D_; g.K = D_; g.K1 = D_;
  g.bias = bo; g.outF = outp; g.ldOut = D_;
  k_gemm_f<<<dim3(D_ / FTN, NTOK / FTM, 1), 256, 0, stream>>>(g);
}
